// Net_5695126634922
// MI455X (gfx1250) — matrix operand order repaired and checked by reading, not compiled, not run on hardware
//
#include <hip/hip_runtime.h>
#include <hip/hip_bf16.h>

#define NN 131072
#define EE 262144
#define HH 256
#define H2C 512
#define LLC 5
#define BB 2048
#define SSC 5
#define VVC 5002
#define VPAD 5056

typedef __attribute__((ext_vector_type(16))) __bf16 v16bf;
typedef __attribute__((ext_vector_type(8)))  float  v8f;
typedef __attribute__((ext_vector_type(4)))  float  v4f;

// ---------------- gather / scatter / elementwise kernels ----------------

__global__ __launch_bounds__(256) void k_embed(
    const int* __restrict__ x, const int* __restrict__ nd,
    const float* __restrict__ ttab, const float* __restrict__ atab,
    const float* __restrict__ dtab, float* __restrict__ out) {
  int n = blockIdx.x, t = threadIdx.x;
  int t0 = x[n * 2], t1 = x[n * 2 + 1], d = nd[n];
  out[(size_t)n * HH + t] = ttab[t0 * HH + t] + atab[t1 * HH + t] + dtab[d * HH + t];
}

__global__ __launch_bounds__(256) void k_vn_init(
    const float* __restrict__ vnw, float* __restrict__ vn) {
  vn[(size_t)blockIdx.x * HH + threadIdx.x] = vnw[threadIdx.x];
}

// h = x_cur + vn[batch];  z = (1+eps)*h   (z then accumulates edge messages)
__global__ __launch_bounds__(256) void k_nodes_pre(
    const float* __restrict__ xc, const float* __restrict__ vn,
    const int* __restrict__ batch, const float* __restrict__ epsArr, int li,
    float* __restrict__ h, float* __restrict__ z) {
  int n = blockIdx.x, t = threadIdx.x;
  float hv = xc[(size_t)n * HH + t] + vn[(size_t)batch[n] * HH + t];
  h[(size_t)n * HH + t] = hv;
  z[(size_t)n * HH + t] = (1.0f + epsArr[li]) * hv;
}

// m = relu(h[src] + etab[ea0] + etab[ea1]);  z[dst] += m  (native f32 atomic)
__global__ __launch_bounds__(256) void k_edges(
    const int* __restrict__ src, const int* __restrict__ dst,
    const int* __restrict__ ea, const float* __restrict__ etab,
    const float* __restrict__ h, float* __restrict__ z) {
  int e = blockIdx.x, t = threadIdx.x;
  float em = etab[ea[e * 2] * HH + t] + etab[ea[e * 2 + 1] * HH + t];
  float m = h[(size_t)src[e] * HH + t] + em;
  m = fmaxf(m, 0.0f);
  unsafeAtomicAdd(&z[(size_t)dst[e] * HH + t], m);
}

__global__ __launch_bounds__(256) void k_copy_bh(
    const float* __restrict__ a, float* __restrict__ b) {
  size_t i = (size_t)blockIdx.x * blockDim.x + threadIdx.x;
  b[i] = a[i];
}

__global__ __launch_bounds__(256) void k_zero_bh(float* __restrict__ p) {
  p[(size_t)blockIdx.x * blockDim.x + threadIdx.x] = 0.0f;
}

__global__ __launch_bounds__(256) void k_segsum(
    const float* __restrict__ xc, const int* __restrict__ batch,
    float* __restrict__ acc) {
  int n = blockIdx.x, t = threadIdx.x;
  unsafeAtomicAdd(&acc[(size_t)batch[n] * HH + t], xc[(size_t)n * HH + t]);
}

// W [K, Nsrc] f32 row-major  ->  Wt [Npad, K] bf16 (transposed, zero-padded rows)
__global__ __launch_bounds__(256) void k_convw(
    const float* __restrict__ W, __bf16* __restrict__ Wt,
    int K, int Nsrc, long total) {
  for (long idx = (long)blockIdx.x * blockDim.x + threadIdx.x; idx < total;
       idx += (long)gridDim.x * blockDim.x) {
    long n = idx / K;
    long k = idx - n * K;
    float v = (n < Nsrc) ? W[(size_t)k * Nsrc + n] : 0.0f;
    Wt[idx] = (__bf16)v;
  }
}

// ---------------- generic WMMA bf16 GEMM ----------------
// C[M,Nout] = epilogue(A[M,K](f32->bf16) x Bt[Npad,K](bf16)^T)
// One wave computes a 32x64 tile (2 m-subtiles x 4 n-tiles); 8 waves/block.
// Fragment layouts used here (wave32; contiguous k, not the ISA table order):
//   A 16x32 bf16: lanes 0-15 row m, elems = K0..15; lanes 16-31 K16..31 (contiguous)
//   B 32x16 bf16: lane n%16 holds 16 CONTIGUOUS K values; lanes 0-15 K0..15, 16-31 K16..31
//   C/D f32:      lane -> col n=lane%16; VGPR r -> row m=(lane/16)*8+r
// mode 0: +bias | mode 1: (+bias)*g+b, relu | mode 2: (+bias)*g+b, prelu(alpha)
__global__ __launch_bounds__(256) void k_gemm(
    const float* __restrict__ A, const __bf16* __restrict__ Bt,
    float* __restrict__ C,
    const float* __restrict__ bias, const float* __restrict__ gamma,
    const float* __restrict__ beta, const float* __restrict__ alphaP,
    int M, int K, int Nout, int mode) {
  int wave = threadIdx.x >> 5;
  int lane = threadIdx.x & 31;
  int m0 = (blockIdx.y * 8 + wave) * 32;
  int n0 = blockIdx.x * 64;
  int lr = lane & 15;
  int hi = lane >> 4;

  const float* arow[2];
  arow[0] = A + (size_t)(m0 + lr) * K;
  arow[1] = A + (size_t)(m0 + 16 + lr) * K;
  v8f acc[2][4] = {};

  for (int k0 = 0; k0 < K; k0 += 32) {
    v16bf a[2];
#pragma unroll
    for (int mi = 0; mi < 2; ++mi) {
      const float* ar = arow[mi];
      const v4f f0 = *(const v4f*)(ar + k0 + hi * 16);
      const v4f f1 = *(const v4f*)(ar + k0 + hi * 16 + 4);
      const v4f f2 = *(const v4f*)(ar + k0 + hi * 16 + 8);
      const v4f f3 = *(const v4f*)(ar + k0 + hi * 16 + 12);
      a[mi][0]  = (__bf16)f0[0]; a[mi][1]  = (__bf16)f0[1];
      a[mi][2]  = (__bf16)f0[2]; a[mi][3]  = (__bf16)f0[3];
      a[mi][4]  = (__bf16)f1[0]; a[mi][5]  = (__bf16)f1[1];
      a[mi][6]  = (__bf16)f1[2]; a[mi][7]  = (__bf16)f1[3];
      a[mi][8]  = (__bf16)f2[0]; a[mi][9]  = (__bf16)f2[1];
      a[mi][10] = (__bf16)f2[2]; a[mi][11] = (__bf16)f2[3];
      a[mi][12] = (__bf16)f3[0]; a[mi][13] = (__bf16)f3[1];
      a[mi][14] = (__bf16)f3[2]; a[mi][15] = (__bf16)f3[3];
    }
#pragma unroll
    for (int t = 0; t < 4; ++t) {
      // contiguous 16 bf16 (32B, aligned) per lane
      const v16bf b = *(const v16bf*)(Bt + (size_t)(n0 + t * 16 + lr) * K + k0 + hi * 16);
      acc[0][t] = __builtin_amdgcn_wmma_f32_16x16x32_bf16(
          false, a[0], false, b, (short)0, acc[0][t], false, false);
      acc[1][t] = __builtin_amdgcn_wmma_f32_16x16x32_bf16(
          false, a[1], false, b, (short)0, acc[1][t], false, false);
    }
  }

  float alpha = (mode == 2) ? alphaP[0] : 0.0f;
#pragma unroll
  for (int t = 0; t < 4; ++t) {
    int col = n0 + t * 16 + lr;
    if (col < Nout) {
      float bi = bias[col];
      float ga = (mode != 0) ? gamma[col] : 1.0f;
      float be = (mode != 0) ? beta[col] : 0.0f;
#pragma unroll
      for (int mi = 0; mi < 2; ++mi) {
#pragma unroll
        for (int r = 0; r < 8; ++r) {
          int row = m0 + mi * 16 + hi * 8 + r;
          float v = acc[mi][t][r] + bi;
          if (mode == 1) { v = v * ga + be; v = fmaxf(v, 0.0f); }
          else if (mode == 2) { v = v * ga + be; v = (v >= 0.0f) ? v : alpha * v; }
          C[(size_t)row * Nout + col] = v;
        }
      }
    }
  }
}

// ---------------- host orchestration ----------------

extern "C" void kernel_launch(void* const* d_in, const int* in_sizes, int n_in,
                              void* d_out, int out_size, void* d_ws, size_t ws_size,
                              hipStream_t stream) {
  const int*   x     = (const int*)d_in[0];
  const int*   nd    = (const int*)d_in[1];
  const int*   eidx  = (const int*)d_in[2];
  const int*   eattr = (const int*)d_in[3];
  const int*   batch = (const int*)d_in[4];
  const float* ttab  = (const float*)d_in[5];
  const float* atab  = (const float*)d_in[6];
  const float* dtab  = (const float*)d_in[7];
  const float* vnw   = (const float*)d_in[8];
  const float* etab  = (const float*)d_in[9];
  const float* eps   = (const float*)d_in[10];
  const float* W1    = (const float*)d_in[11];
  const float* b1    = (const float*)d_in[12];
  const float* g1    = (const float*)d_in[13];
  const float* be1   = (const float*)d_in[14];
  const float* W2    = (const float*)d_in[15];
  const float* b2    = (const float*)d_in[16];
  const float* Vw1   = (const float*)d_in[17];
  const float* Vb1   = (const float*)d_in[18];
  const float* Vg1   = (const float*)d_in[19];
  const float* Vbe1  = (const float*)d_in[20];
  const float* Va1   = (const float*)d_in[21];
  const float* Vw2   = (const float*)d_in[22];
  const float* Vb2   = (const float*)d_in[23];
  const float* Vg2   = (const float*)d_in[24];
  const float* Vbe2  = (const float*)d_in[25];
  const float* Va2   = (const float*)d_in[26];
  const float* Wp    = (const float*)d_in[27];
  const float* bp    = (const float*)d_in[28];
  float* out = (float*)d_out;

  char* ws = (char*)d_ws;
  size_t off = 0;
  auto alloc = [&](size_t bytes) -> void* {
    void* p = ws + off;
    off += (bytes + 255) & ~(size_t)255;
    return p;
  };
  float*  h    = (float*)alloc((size_t)NN * HH * 4);
  float*  z    = (float*)alloc((size_t)NN * HH * 4);
  float*  z2   = (float*)alloc((size_t)NN * H2C * 4);
  float*  xa   = (float*)alloc((size_t)NN * HH * 4);
  float*  xb   = (float*)alloc((size_t)NN * HH * 4);
  float*  vn   = (float*)alloc((size_t)BB * HH * 4);
  float*  vt   = (float*)alloc((size_t)BB * HH * 4);
  float*  u    = (float*)alloc((size_t)BB * H2C * 4);
  float*  hg   = (float*)alloc((size_t)BB * HH * 4);
  __bf16* W1t  = (__bf16*)alloc((size_t)LLC * H2C * HH * 2);
  __bf16* W2t  = (__bf16*)alloc((size_t)LLC * HH * H2C * 2);
  __bf16* Vw1t = (__bf16*)alloc((size_t)(LLC - 1) * H2C * HH * 2);
  __bf16* Vw2t = (__bf16*)alloc((size_t)(LLC - 1) * HH * H2C * 2);
  __bf16* Wpt  = (__bf16*)alloc((size_t)SSC * VPAD * HH * 2);

  // ---- weight conversion (f32 [K,N] -> transposed bf16 [Npad,K]) ----
  const long totHW = (long)H2C * HH;  // 512*256 both directions
  for (int i = 0; i < LLC; i++) {
    k_convw<<<(totHW + 255) / 256, 256, 0, stream>>>(
        W1 + (size_t)i * HH * H2C, W1t + (size_t)i * H2C * HH, HH, H2C, totHW);
    k_convw<<<(totHW + 255) / 256, 256, 0, stream>>>(
        W2 + (size_t)i * H2C * HH, W2t + (size_t)i * HH * H2C, H2C, HH, totHW);
  }
  for (int i = 0; i < LLC - 1; i++) {
    k_convw<<<(totHW + 255) / 256, 256, 0, stream>>>(
        Vw1 + (size_t)i * HH * H2C, Vw1t + (size_t)i * H2C * HH, HH, H2C, totHW);
    k_convw<<<(totHW + 255) / 256, 256, 0, stream>>>(
        Vw2 + (size_t)i * H2C * HH, Vw2t + (size_t)i * HH * H2C, H2C, HH, totHW);
  }
  const long totWp = (long)VPAD * HH;
  for (int s = 0; s < SSC; s++) {
    k_convw<<<(totWp + 255) / 256, 256, 0, stream>>>(
        Wp + (size_t)s * HH * VVC, Wpt + (size_t)s * VPAD * HH, HH, VVC, totWp);
  }

  // ---- embeddings + VN init ----
  k_embed<<<NN, HH, 0, stream>>>(x, nd, ttab, atab, dtab, xa);
  k_vn_init<<<BB, HH, 0, stream>>>(vnw, vn);

  const int* src = eidx;
  const int* dst = eidx + EE;

  float* xc = xa;
  float* xn = xb;
  for (int i = 0; i < LLC; i++) {
    k_nodes_pre<<<NN, HH, 0, stream>>>(xc, vn, batch, eps, i, h, z);
    k_edges<<<EE, HH, 0, stream>>>(src, dst, eattr, etab + (size_t)i * 8 * HH, h, z);

    {  // z @ W1 -> z2, BN + ReLU
      dim3 g(H2C / 64, NN / 256);
      k_gemm<<<g, 256, 0, stream>>>(z, W1t + (size_t)i * H2C * HH, z2,
                                    b1 + (size_t)i * H2C, g1 + (size_t)i * H2C,
                                    be1 + (size_t)i * H2C, nullptr,
                                    NN, HH, H2C, 1);
    }
    {  // z2 @ W2 -> x_new, bias
      dim3 g(HH / 64, NN / 256);
      k_gemm<<<g, 256, 0, stream>>>(z2, W2t + (size_t)i * HH * H2C, xn,
                                    b2 + (size_t)i * HH, nullptr, nullptr, nullptr,
                                    NN, H2C, HH, 0);
    }
    if (i < LLC - 1) {  // virtual node MLP
      k_copy_bh<<<(BB * HH) / 256, 256, 0, stream>>>(vn, vt);
      k_segsum<<<NN, HH, 0, stream>>>(xc, batch, vt);
      dim3 ga(H2C / 64, BB / 256);
      k_gemm<<<ga, 256, 0, stream>>>(vt, Vw1t + (size_t)i * H2C * HH, u,
                                     Vb1 + (size_t)i * H2C, Vg1 + (size_t)i * H2C,
                                     Vbe1 + (size_t)i * H2C, Va1 + i,
                                     BB, HH, H2C, 2);
      dim3 gb(HH / 64, BB / 256);
      k_gemm<<<gb, 256, 0, stream>>>(u, Vw2t + (size_t)i * HH * H2C, vn,
                                     Vb2 + (size_t)i * HH, Vg2 + (size_t)i * HH,
                                     Vbe2 + (size_t)i * HH, Va2 + i,
                                     BB, H2C, HH, 2);
    }
    float* tmp = xc; xc = xn; xn = tmp;
  }

  // ---- global add pool + prediction head (T = 1.0) ----
  k_zero_bh<<<(BB * HH) / 256, 256, 0, stream>>>(hg);
  k_segsum<<<NN, HH, 0, stream>>>(xc, batch, hg);
  for (int s = 0; s < SSC; s++) {
    dim3 g(VPAD / 64, BB / 256);
    k_gemm<<<g, 256, 0, stream>>>(hg, Wpt + (size_t)s * VPAD * HH,
                                  out + (size_t)s * BB * VVC,
                                  bp + (size_t)s * VVC, nullptr, nullptr, nullptr,
                                  BB, HH, VVC, 0);
  }
}
